// TemporalAttention_38036230374021
// MI455X (gfx1250) — compile-verified, matrix operand order checked by reading, not run on hardware
//
#include <hip/hip_runtime.h>

// Problem constants (from reference)
constexpr int Bn = 4, Tn = 2048, Cn = 512, Hn = 8, Dn = 64;
constexpr int Mtot = Bn * Tn;        // 8192 rows for both GEMMs

typedef __attribute__((ext_vector_type(16))) _Float16 v16h;
typedef __attribute__((ext_vector_type(8)))  float    v8f;
typedef __attribute__((ext_vector_type(4)))  unsigned int v4u;
typedef __attribute__((ext_vector_type(8)))  int      v8i;
typedef __attribute__((ext_vector_type(4)))  int      v4i;

#if defined(__has_builtin)
#  if __has_builtin(__builtin_amdgcn_tensor_load_to_lds)
#    define HAVE_TDM 1
#  endif
#endif
#ifndef HAVE_TDM
#  define HAVE_TDM 0
#endif

// ---------------------------------------------------------------------------
// WMMA helpers (CDNA5 16x16x32 f16 -> f32)
// ---------------------------------------------------------------------------
__device__ __forceinline__ v8f wmma_f16(v16h a, v16h b, v8f c) {
    return __builtin_amdgcn_wmma_f32_16x16x32_f16(false, a, false, b,
                                                  (short)0, c, false, false);
}

// K-offset for element-pair v (0..7) of a 16-bit A/B fragment (ISA 7.12.2)
__device__ __forceinline__ int frag_koff(int v, int hi) {
    return ((v & 4) << 2) + ((v & 3) << 1) + (hi << 3);
}

// Load a 16x32 fragment whose rows (M for A, N for B) have contiguous K.
__device__ __forceinline__ v16h load_frag_rowk(const _Float16* base, int ld, int lane) {
    const int r = lane & 15, hi = lane >> 4;
    v16h f;
#pragma unroll
    for (int v = 0; v < 8; ++v) {
        const _Float16* p = base + r * ld + frag_koff(v, hi);
        f[2 * v]     = p[0];
        f[2 * v + 1] = p[1];
    }
    return f;
}

__device__ __forceinline__ v8f zero8() {
    v8f z = {0.f, 0.f, 0.f, 0.f, 0.f, 0.f, 0.f, 0.f};
    return z;
}

// ---------------------------------------------------------------------------
// Tensor Data Mover: 2D f16 tile DMA global -> LDS (D# per ISA 08 §8)
// tile_w elements contiguous, tile_h rows, row stride in elements.
// ---------------------------------------------------------------------------
__device__ __forceinline__ unsigned lds_off(const void* p) {
    // generic LDS pointer: low 32 bits are the LDS byte offset
    return (unsigned)(unsigned long long)(uintptr_t)p;
}

#if HAVE_TDM
__device__ __forceinline__ void tdm_load_2d_f16(unsigned lds_byte, const _Float16* gsrc,
                                                int tile_w, int tile_h, int row_stride) {
    unsigned long long ga = (unsigned long long)(uintptr_t)gsrc;
    v4u g0;
    g0[0] = 1u;                                              // count=1, no gather
    g0[1] = lds_byte;                                        // lds_addr
    g0[2] = (unsigned)(ga & 0xffffffffu);                    // global_addr[31:0]
    g0[3] = (unsigned)((ga >> 32) & 0x01ffffffu) | (2u << 30); // addr[56:32] | type=2
    v8i g1;
    g1[0] = (int)(1u << 16);                                 // data_size=2B, mask=0
    g1[1] = (int)(((unsigned)tile_w & 0xffffu) << 16);       // tensor_dim0 lo16
    g1[2] = (int)((((unsigned)tile_w >> 16) & 0xffffu) |
                  (((unsigned)tile_h & 0xffffu) << 16));     // dim0 hi16 | dim1 lo16
    g1[3] = (int)((((unsigned)tile_h >> 16) & 0xffffu) |
                  (((unsigned)tile_w & 0xffffu) << 16));     // dim1 hi16 | tile_dim0
    g1[4] = (int)((unsigned)tile_h & 0xffffu);               // tile_dim1 | tile_dim2=0
    g1[5] = (int)(unsigned)row_stride;                       // tensor_dim0_stride lo32
    g1[6] = 0;
    g1[7] = 0;
    v4i z4 = {0, 0, 0, 0};
#if __clang_major__ <= 22
    __builtin_amdgcn_tensor_load_to_lds(g0, g1, z4, z4, 0);
#else
    v8i z8 = {0, 0, 0, 0, 0, 0, 0, 0};
    __builtin_amdgcn_tensor_load_to_lds(g0, g1, z4, z4, z8, 0);
#endif
}
#endif

// Fallback / general staging: 128x64 f16 tile via b128 copies (256 threads)
__device__ __forceinline__ void stage128x64(_Float16* dst, const _Float16* src,
                                            int tid, int src_ld) {
#pragma unroll
    for (int j = 0; j < 4; ++j) {
        int v = tid + j * 256;
        int row = v >> 3, cv = (v & 7) * 8;
        *(uint4*)(dst + row * 64 + cv) = *(const uint4*)(src + (size_t)row * src_ld + cv);
    }
}

// ---------------------------------------------------------------------------
// Stage 0: fp32 -> f16 conversion
// ---------------------------------------------------------------------------
__global__ void k_cvt_f16(const float* __restrict__ in, _Float16* __restrict__ out, int n) {
    int i = blockIdx.x * blockDim.x + threadIdx.x;
    if (i < n) out[i] = (_Float16)in[i];
}

// ---------------------------------------------------------------------------
// Shared GEMM mainloop: block tile 128(M) x 128(N), K staged 64 deep,
// double-buffered TDM DMA.  8 waves, each computing 32x64 (2x4 WMMA tiles).
// A: [M,512] row-major f16.  W: [N,512] row-major f16 (B = W^T).
// ---------------------------------------------------------------------------
__device__ __forceinline__ void gemm_mainloop(
    const _Float16* __restrict__ A, const _Float16* __restrict__ W,
    _Float16* sA, _Float16* sB, int m0, int n0, int tid, v8f acc[2][4])
{
    const int lane = tid & 31, wave = tid >> 5;
    const int wm = (wave >> 1) * 32, wn = (wave & 1) * 64;
    constexpr int TILE = 128 * 64;

    // Prologue: stage k0 = 0 into buffer 0
#if HAVE_TDM
    if (wave == 0) {
        tdm_load_2d_f16(lds_off(sA), A + (size_t)m0 * Cn, 64, 128, Cn);
        tdm_load_2d_f16(lds_off(sB), W + (size_t)n0 * Cn, 64, 128, Cn);
        __builtin_amdgcn_s_wait_tensorcnt(0);
    }
#else
    stage128x64(sA, A + (size_t)m0 * Cn, tid, Cn);
    stage128x64(sB, W + (size_t)n0 * Cn, tid, Cn);
#endif
    __syncthreads();

    for (int k0 = 0; k0 < Cn; k0 += 64) {
        const int cur = (k0 >> 6) & 1, nxt = cur ^ 1;
        _Float16* cA = sA + cur * TILE;
        _Float16* cB = sB + cur * TILE;
        const bool has_next = (k0 + 64) < Cn;

        // Kick off DMA of the next K-slab while we compute on this one
        if (has_next) {
#if HAVE_TDM
            if (wave == 0) {
                tdm_load_2d_f16(lds_off(sA + nxt * TILE),
                                A + (size_t)m0 * Cn + (k0 + 64), 64, 128, Cn);
                tdm_load_2d_f16(lds_off(sB + nxt * TILE),
                                W + (size_t)n0 * Cn + (k0 + 64), 64, 128, Cn);
            }
#else
            stage128x64(sA + nxt * TILE, A + (size_t)m0 * Cn + (k0 + 64), tid, Cn);
            stage128x64(sB + nxt * TILE, W + (size_t)n0 * Cn + (k0 + 64), tid, Cn);
#endif
        }

#pragma unroll
        for (int ks = 0; ks < 2; ++ks) {
            v16h a0 = load_frag_rowk(cA + (wm + 0)  * 64 + ks * 32, 64, lane);
            v16h a1 = load_frag_rowk(cA + (wm + 16) * 64 + ks * 32, 64, lane);
#pragma unroll
            for (int jn = 0; jn < 4; ++jn) {
                v16h b = load_frag_rowk(cB + (wn + jn * 16) * 64 + ks * 32, 64, lane);
                acc[0][jn] = wmma_f16(a0, b, acc[0][jn]);
                acc[1][jn] = wmma_f16(a1, b, acc[1][jn]);
            }
        }

#if HAVE_TDM
        if (wave == 0) __builtin_amdgcn_s_wait_tensorcnt(0);
#endif
        __syncthreads();
    }
}

// ---------------------------------------------------------------------------
// Stage 1: QKV GEMM.  out[m,n] = x.W^T + bias, scattered into [B,H,T,D] f16
// ---------------------------------------------------------------------------
__global__ __launch_bounds__(256) void k_qkv_gemm(
    const _Float16* __restrict__ A, const _Float16* __restrict__ W,
    const float* __restrict__ bias,
    _Float16* __restrict__ Qo, _Float16* __restrict__ Ko, _Float16* __restrict__ Vo)
{
    __shared__ __align__(16) _Float16 sA[2 * 128 * 64];
    __shared__ __align__(16) _Float16 sB[2 * 128 * 64];

    const int tid = threadIdx.x;
    const int lane = tid & 31, wave = tid >> 5;
    const int m0 = blockIdx.x * 128, n0 = blockIdx.y * 128;
    const int wm = (wave >> 1) * 32, wn = (wave & 1) * 64;

    v8f acc[2][4];
#pragma unroll
    for (int i = 0; i < 2; ++i)
#pragma unroll
        for (int j = 0; j < 4; ++j) acc[i][j] = zero8();

    gemm_mainloop(A, W, sA, sB, m0, n0, tid, acc);

    const int hi = lane >> 4, nl = lane & 15;
#pragma unroll
    for (int i = 0; i < 2; ++i)
#pragma unroll
        for (int jn = 0; jn < 4; ++jn)
#pragma unroll
            for (int r = 0; r < 8; ++r) {
                int mg = m0 + wm + i * 16 + r + hi * 8;
                int ng = n0 + wn + jn * 16 + nl;
                float val = acc[i][jn][r] + bias[ng];
                int b = mg >> 11, t = mg & (Tn - 1);
                int which = ng >> 9, c = ng & (Cn - 1);
                int h = c >> 6, d = c & (Dn - 1);
                _Float16* dst = (which == 0) ? Qo : (which == 1) ? Ko : Vo;
                dst[(((size_t)(b * Hn + h)) * Tn + t) * Dn + d] = (_Float16)val;
            }
}

// ---------------------------------------------------------------------------
// Stage 2: flash attention. One block per (bh, 64-row q tile), 4 waves,
// each wave owns 16 q rows.  K tile DMA'd by TDM; V tile staged TRANSPOSED
// so every fragment read in the streaming loop is contiguous ds_load_b128.
// ---------------------------------------------------------------------------
__global__ __launch_bounds__(128) void k_flash_attn(
    const _Float16* __restrict__ Q, const _Float16* __restrict__ K,
    const _Float16* __restrict__ V, _Float16* __restrict__ O)
{
    __shared__ __align__(16) _Float16 sK[64 * 64];     // [kv][d]
    __shared__ __align__(16) _Float16 sVt[64 * 64];    // [d][kv]  (transposed)
    __shared__ __align__(16) _Float16 sP[4 * 16 * 64]; // per-wave P strips

    const int tid = threadIdx.x, lane = tid & 31, wave = tid >> 5;
    const int bh = blockIdx.y;
    const int q0 = blockIdx.x * 64;
    const int b = bh >> 3, h = bh & (Hn - 1);
    const int m = lane & 15, hi = lane >> 4;

    const _Float16* Qb = Q + (size_t)bh * Tn * Dn;
    const _Float16* Kb = K + (size_t)bh * Tn * Dn;
    const _Float16* Vb = V + (size_t)bh * Tn * Dn;

    // This wave's Q fragments (16 rows x 64 d), 2 k-steps of 32
    v16h qf[2];
#pragma unroll
    for (int ks = 0; ks < 2; ++ks) {
#pragma unroll
        for (int v = 0; v < 8; ++v) {
            int d = frag_koff(v, hi) + ks * 32;
            const _Float16* p = Qb + (size_t)(q0 + wave * 16 + m) * Dn + d;
            qf[ks][2 * v]     = p[0];
            qf[ks][2 * v + 1] = p[1];
        }
    }

    float row_max[8], row_sum[8];
#pragma unroll
    for (int r = 0; r < 8; ++r) { row_max[r] = -1e30f; row_sum[r] = 0.f; }
    v8f o[4];
#pragma unroll
    for (int nt = 0; nt < 4; ++nt) o[nt] = zero8();

    for (int kv0 = 0; kv0 < Tn; kv0 += 64) {
        __syncthreads();   // previous iteration's LDS reads complete

        // K tile via TDM (row-major, contiguous)
#if HAVE_TDM
        if (wave == 0)
            tdm_load_2d_f16(lds_off(sK), Kb + (size_t)kv0 * Dn, 64, 64, 64);
#else
#pragma unroll
        for (int j = 0; j < 4; ++j) {
            int v = tid + j * 128;
            int row = v >> 3, cv = (v & 7) * 8;
            *(uint4*)(sK + row * 64 + cv) =
                *(const uint4*)(Kb + (size_t)(kv0 + row) * Dn + cv);
        }
#endif
        // V tile: thread-staged with transpose -> sVt[d][kv]
#pragma unroll
        for (int j = 0; j < 4; ++j) {
            int v = tid + j * 128;
            int row = v >> 3, cv = (v & 7) * 8;
            uint4 tmp = *(const uint4*)(Vb + (size_t)(kv0 + row) * Dn + cv);
            const _Float16* tf = (const _Float16*)&tmp;
#pragma unroll
            for (int e = 0; e < 8; ++e) sVt[(cv + e) * 64 + row] = tf[e];
        }
#if HAVE_TDM
        if (wave == 0) __builtin_amdgcn_s_wait_tensorcnt(0);
#endif
        __syncthreads();

        // S = (Q . K^T) * 1/sqrt(D)
        v8f s[4];
#pragma unroll
        for (int nt = 0; nt < 4; ++nt) {
            v8f sa = zero8();
#pragma unroll
            for (int ks = 0; ks < 2; ++ks) {
                v16h bf = load_frag_rowk(sK + (nt * 16) * 64 + ks * 32, 64, lane);
                sa = wmma_f16(qf[ks], bf, sa);
            }
#pragma unroll
            for (int r = 0; r < 8; ++r) s[nt][r] = sa[r] * 0.125f;
        }

        // Online softmax (rows r+hi*8; xor masks 1,2,4,8 stay inside a half)
#pragma unroll
        for (int r = 0; r < 8; ++r) {
            float mx = fmaxf(fmaxf(s[0][r], s[1][r]), fmaxf(s[2][r], s[3][r]));
            mx = fmaxf(mx, __shfl_xor(mx, 1, 32));
            mx = fmaxf(mx, __shfl_xor(mx, 2, 32));
            mx = fmaxf(mx, __shfl_xor(mx, 4, 32));
            mx = fmaxf(mx, __shfl_xor(mx, 8, 32));
            float nm = fmaxf(row_max[r], mx);
            float corr = __expf(row_max[r] - nm);
            row_max[r] = nm;
            float ps = 0.f;
#pragma unroll
            for (int nt = 0; nt < 4; ++nt) {
                float p = __expf(s[nt][r] - nm);
                s[nt][r] = p;
                ps += p;
            }
            ps += __shfl_xor(ps, 1, 32);
            ps += __shfl_xor(ps, 2, 32);
            ps += __shfl_xor(ps, 4, 32);
            ps += __shfl_xor(ps, 8, 32);
            row_sum[r] = row_sum[r] * corr + ps;
#pragma unroll
            for (int nt = 0; nt < 4; ++nt) o[nt][r] *= corr;
        }

        // P -> LDS f16 (wave-private strip), reload as A-fragments
        _Float16* wP = sP + wave * (16 * 64);
#pragma unroll
        for (int nt = 0; nt < 4; ++nt)
#pragma unroll
            for (int r = 0; r < 8; ++r)
                wP[(r + hi * 8) * 64 + nt * 16 + (lane & 15)] = (_Float16)s[nt][r];
        __syncthreads();

        // O += P . V  (B-frag rows = d of sVt, contiguous kv)
#pragma unroll
        for (int ks = 0; ks < 2; ++ks) {
            v16h pf;
#pragma unroll
            for (int v = 0; v < 8; ++v) {
                const _Float16* p = wP + m * 64 + frag_koff(v, hi) + ks * 32;
                pf[2 * v]     = p[0];
                pf[2 * v + 1] = p[1];
            }
#pragma unroll
            for (int nt = 0; nt < 4; ++nt) {
                v16h vf = load_frag_rowk(sVt + (nt * 16) * 64 + ks * 32, 64, lane);
                o[nt] = wmma_f16(pf, vf, o[nt]);
            }
        }
    }

    // Normalize and store to [B,T,C] with c = h*64 + d
#pragma unroll
    for (int nt = 0; nt < 4; ++nt)
#pragma unroll
        for (int r = 0; r < 8; ++r) {
            float val = o[nt][r] / row_sum[r];
            int t = q0 + wave * 16 + r + hi * 8;
            int d = nt * 16 + (lane & 15);
            O[((size_t)b * Tn + t) * Cn + h * Dn + d] = (_Float16)val;
        }
}

// ---------------------------------------------------------------------------
// Stage 3: output projection -> fp32 d_out + bias
// ---------------------------------------------------------------------------
__global__ __launch_bounds__(256) void k_proj_gemm(
    const _Float16* __restrict__ A, const _Float16* __restrict__ W,
    const float* __restrict__ bias, float* __restrict__ out)
{
    __shared__ __align__(16) _Float16 sA[2 * 128 * 64];
    __shared__ __align__(16) _Float16 sB[2 * 128 * 64];

    const int tid = threadIdx.x;
    const int lane = tid & 31, wave = tid >> 5;
    const int m0 = blockIdx.x * 128, n0 = blockIdx.y * 128;
    const int wm = (wave >> 1) * 32, wn = (wave & 1) * 64;

    v8f acc[2][4];
#pragma unroll
    for (int i = 0; i < 2; ++i)
#pragma unroll
        for (int j = 0; j < 4; ++j) acc[i][j] = zero8();

    gemm_mainloop(A, W, sA, sB, m0, n0, tid, acc);

    const int hi = lane >> 4, nl = lane & 15;
#pragma unroll
    for (int i = 0; i < 2; ++i)
#pragma unroll
        for (int jn = 0; jn < 4; ++jn)
#pragma unroll
            for (int r = 0; r < 8; ++r) {
                int mg = m0 + wm + i * 16 + r + hi * 8;
                int ng = n0 + wn + jn * 16 + nl;
                out[(size_t)mg * Cn + ng] = acc[i][jn][r] + bias[ng];
            }
}

// ---------------------------------------------------------------------------
// Host-side launch
// ---------------------------------------------------------------------------
extern "C" void kernel_launch(void* const* d_in, const int* in_sizes, int n_in,
                              void* d_out, int out_size, void* d_ws, size_t ws_size,
                              hipStream_t stream) {
    const float* x      = (const float*)d_in[0];   // [B,T,C]
    const float* qkv_w  = (const float*)d_in[1];   // [3C,C]
    const float* qkv_b  = (const float*)d_in[2];   // [3C]
    const float* proj_w = (const float*)d_in[3];   // [C,C]
    const float* proj_b = (const float*)d_in[4];   // [C]
    float* out = (float*)d_out;                    // [B,T,C] fp32

    const size_t nBTC = (size_t)Bn * Tn * Cn;
    const size_t nW1  = (size_t)3 * Cn * Cn;
    const size_t nW2  = (size_t)Cn * Cn;
    const size_t nQKV = (size_t)Bn * Hn * Tn * Dn;

    char* ws = (char*)d_ws;
    _Float16* xh  = (_Float16*)ws;  ws += nBTC * sizeof(_Float16);
    _Float16* w1h = (_Float16*)ws;  ws += nW1  * sizeof(_Float16);
    _Float16* w2h = (_Float16*)ws;  ws += nW2  * sizeof(_Float16);
    _Float16* Qh  = (_Float16*)ws;  ws += nQKV * sizeof(_Float16);
    _Float16* Kh  = (_Float16*)ws;  ws += nQKV * sizeof(_Float16);
    _Float16* Vh  = (_Float16*)ws;  ws += nQKV * sizeof(_Float16);
    _Float16* Oh  = (_Float16*)ws;  ws += nBTC * sizeof(_Float16);

    k_cvt_f16<<<(int)((nBTC + 255) / 256), 256, 0, stream>>>(x,      xh,  (int)nBTC);
    k_cvt_f16<<<(int)((nW1  + 255) / 256), 256, 0, stream>>>(qkv_w,  w1h, (int)nW1);
    k_cvt_f16<<<(int)((nW2  + 255) / 256), 256, 0, stream>>>(proj_w, w2h, (int)nW2);

    // QKV GEMM (M=8192, N=1536, K=512) -> Q/K/V [B,H,T,D] f16
    k_qkv_gemm<<<dim3(Mtot / 128, (3 * Cn) / 128), 256, 0, stream>>>(
        xh, w1h, qkv_b, Qh, Kh, Vh);

    // Flash attention -> Oh [B,T,C] f16
    k_flash_attn<<<dim3(Tn / 64, Bn * Hn), 128, 0, stream>>>(Qh, Kh, Vh, Oh);

    // Projection (M=8192, N=512, K=512) -> fp32 out + bias
    k_proj_gemm<<<dim3(Mtot / 128, Cn / 128), 256, 0, stream>>>(Oh, w2h, proj_b, out);
}
